// PathOptimizer_70377334112306
// MI455X (gfx1250) — compile-verified, matrix operand order checked by reading, not run on hardware
//
#include <hip/hip_runtime.h>
#include <cmath>

// ---------------------------------------------------------------------------
// PathOptimizer pipeline for gfx1250 (MI455X), wave32, WMMA f16->f32.
// All dense layers use v_wmma_f32_16x16x32_f16 with branch-free 128-bit
// fragment loads (K padded to 32 where needed, so no K-tail path exists).
// LSTM recurrence is a persistent single-workgroup kernel with
// register-resident cell state aligned to the WMMA D-fragment layout,
// f16 h-state in LDS, and global_prefetch of the next timestep's inputs.
// ---------------------------------------------------------------------------

typedef __attribute__((ext_vector_type(16))) _Float16 v16h;
typedef __attribute__((ext_vector_type(8)))  _Float16 v8h;
typedef __attribute__((ext_vector_type(4)))  _Float16 v4h;
typedef __attribute__((ext_vector_type(8)))  float    v8f;
typedef __attribute__((ext_vector_type(4)))  float    v4f;

#define B_   64
#define T_   64
#define HID_ 256
#define EMB_ 768
#define L_   32
#define H_   128
#define W_   128
#define CA_  16

__device__ __forceinline__ float gelu_f(float x) {
    return 0.5f * x * (1.0f + erff(x * 0.70710678118654752f));
}
__device__ __forceinline__ float sig_f(float x) {
    return 1.0f / (1.0f + expf(-x));
}
// 16-bit A/B operand per-lane K layout for 16x16x32 (ISA 7.12.2):
// lanes 0-15 hold K {0..7,16..23}; lanes 16-31 hold K {8..15,24..31}.
// Each lane's fragment is therefore two contiguous 8-element runs.
__device__ __forceinline__ v16h pack_f32x16(v4f a0, v4f a1, v4f a2, v4f a3) {
    v4h h0 = __builtin_convertvector(a0, v4h);
    v4h h1 = __builtin_convertvector(a1, v4h);
    v4h h2 = __builtin_convertvector(a2, v4h);
    v4h h3 = __builtin_convertvector(a3, v4h);
    v8h lo = __builtin_shufflevector(h0, h1, 0, 1, 2, 3, 4, 5, 6, 7);
    v8h hi = __builtin_shufflevector(h2, h3, 0, 1, 2, 3, 4, 5, 6, 7);
    return __builtin_shufflevector(lo, hi, 0, 1, 2, 3, 4, 5, 6, 7,
                                   8, 9, 10, 11, 12, 13, 14, 15);
}
__device__ __forceinline__ v16h cat_h8(v8h lo, v8h hi) {
    return __builtin_shufflevector(lo, hi, 0, 1, 2, 3, 4, 5, 6, 7,
                                   8, 9, 10, 11, 12, 13, 14, 15);
}

// ---------------------------------------------------------------------------
// Generic WMMA GEMM: Y(M,N) = act( X(M,K) @ W(N,K)^T + b1 + b2 )
// One wave per 16x16 output tile. REQUIRES M % 16 == 0 and K % 32 == 0
// (guaranteed by construction; K=3 layers are zero-padded to 32 up front).
// N-full case (N % 16 == 0, wave-uniform): pure 128-bit loads, no masks.
// N-tail case: clamped addresses + value masks, never exec branches.
// act: 0=identity, 1=gelu(exact), 2=10*tanh, 3=sigmoid
// ---------------------------------------------------------------------------
__global__ void wmma_gemm_kernel(const float* __restrict__ X,
                                 const float* __restrict__ Wt,
                                 const float* __restrict__ b1,
                                 const float* __restrict__ b2,
                                 float* __restrict__ Y,
                                 int M, int N, int K, int act)
{
    const int ntn    = (N + 15) >> 4;
    const int nwaves = (M >> 4) * ntn;
    const int wave   = blockIdx.x * (blockDim.x >> 5) + (threadIdx.x >> 5);
    if (wave >= nwaves) return;                  // wave-uniform exit
    const int tm   = wave / ntn;
    const int tn   = wave % ntn;
    const int lane = threadIdx.x & 31;
    const int l16  = lane & 15;
    const int hsel = lane >> 4;
    const int mrow = tm * 16 + l16;
    const int ncol = tn * 16 + l16;

    v8f acc = {};
    const float* pa = X + (size_t)mrow * K + 8 * hsel;
    if ((N & 15) == 0) {
        // Common case: no masking anywhere in the K loop.
        const float* pb = Wt + (size_t)ncol * K + 8 * hsel;
        for (int k0 = 0; k0 < K; k0 += 32) {
            v4f a0 = *(const v4f*)(pa + k0);
            v4f a1 = *(const v4f*)(pa + k0 + 4);
            v4f a2 = *(const v4f*)(pa + k0 + 16);
            v4f a3 = *(const v4f*)(pa + k0 + 20);
            v4f c0 = *(const v4f*)(pb + k0);
            v4f c1 = *(const v4f*)(pb + k0 + 4);
            v4f c2 = *(const v4f*)(pb + k0 + 16);
            v4f c3 = *(const v4f*)(pb + k0 + 20);
            v16h a = pack_f32x16(a0, a1, a2, a3);
            v16h b = pack_f32x16(c0, c1, c2, c3);
            acc = __builtin_amdgcn_wmma_f32_16x16x32_f16(false, a, false, b,
                                                         (short)0, acc, false, false);
        }
    } else {
        // N-tail (score head, N=1): clamped rows + value masks (v_cndmask).
        const bool  nv     = (ncol < N);
        const int   ncl    = nv ? ncol : (N - 1);
        const float bscale = nv ? 1.0f : 0.0f;
        const float* pb = Wt + (size_t)ncl * K + 8 * hsel;
        for (int k0 = 0; k0 < K; k0 += 32) {
            v4f a0 = *(const v4f*)(pa + k0);
            v4f a1 = *(const v4f*)(pa + k0 + 4);
            v4f a2 = *(const v4f*)(pa + k0 + 16);
            v4f a3 = *(const v4f*)(pa + k0 + 20);
            v4f c0 = *(const v4f*)(pb + k0)      * bscale;
            v4f c1 = *(const v4f*)(pb + k0 + 4)  * bscale;
            v4f c2 = *(const v4f*)(pb + k0 + 16) * bscale;
            v4f c3 = *(const v4f*)(pb + k0 + 20) * bscale;
            v16h a = pack_f32x16(a0, a1, a2, a3);
            v16h b = pack_f32x16(c0, c1, c2, c3);
            acc = __builtin_amdgcn_wmma_f32_16x16x32_f16(false, a, false, b,
                                                         (short)0, acc, false, false);
        }
    }
    #pragma unroll
    for (int r = 0; r < 8; ++r) {
        int m = tm * 16 + r + 8 * hsel;          // D layout: M = r + 8*(lane>=16)
        int n = tn * 16 + l16;
        if (n < N) {
            float v = acc[r];
            if (b1) v += b1[n];
            if (b2) v += b2[n];
            if      (act == 1) v = gelu_f(v);
            else if (act == 2) v = 10.0f * tanhf(v);
            else if (act == 3) v = sig_f(v);
            Y[(size_t)m * N + n] = v;
        }
    }
}

// ---------------------------------------------------------------------------
// Small elementwise / reduction / padding kernels
// ---------------------------------------------------------------------------
__global__ void mean_l_kernel(const float* __restrict__ gf, float* __restrict__ out)
{   // (B,L,EMB) -> (B,EMB) mean over L
    int idx = blockIdx.x * blockDim.x + threadIdx.x;
    if (idx >= B_ * EMB_) return;
    int b = idx / EMB_, e = idx % EMB_;
    float s = 0.f;
    for (int l = 0; l < L_; ++l) s += gf[((size_t)(b * L_ + l)) * EMB_ + e];
    out[idx] = s * (1.0f / L_);
}

__global__ void cvt_f16_kernel(const float* __restrict__ in, _Float16* __restrict__ out, int n)
{
    int idx = blockIdx.x * blockDim.x + threadIdx.x;
    if (idx < n) out[idx] = (_Float16)in[idx];
}

__global__ void pad_k_kernel(const float* __restrict__ in,   // (rows, kin)
                             float* __restrict__ out,        // (rows, 32) zero-padded
                             int rows, int kin)
{
    int idx = blockIdx.x * blockDim.x + threadIdx.x;
    if (idx >= rows * 32) return;
    int r = idx >> 5, k = idx & 31;
    out[idx] = (k < kin) ? in[r * kin + k] : 0.0f;
}

// ---------------------------------------------------------------------------
// Geodesic refinement: 50 momentum-FD steps. One block per batch, thread = t.
// ---------------------------------------------------------------------------
__global__ void geodesic_kernel(const float* __restrict__ start,
                                const float* __restrict__ genc,
                                const float* __restrict__ step_p,
                                const float* __restrict__ mom_p,
                                float* __restrict__ path)
{
    __shared__ float P[T_][3];
    const int b = blockIdx.x, t = threadIdx.x;
    const float step = step_p[0], mom = mom_p[0];
    const float alpha = (float)t / (float)(T_ - 1);
    float p[3], v[3] = {0.f, 0.f, 0.f};
    for (int d = 0; d < 3; ++d) {
        float s0 = start[b * 3 + d];
        float g0 = genc[b * HID_ + d];           // goal_pos = goal_enc[:, :3]
        p[d] = s0 * (1.f - alpha) + g0 * alpha;
        P[t][d] = p[d];
    }
    __syncthreads();
    const float eps = 1e-4f;
    const bool interior = (t > 0) && (t < T_ - 1);
    for (int it = 0; it < 50; ++it) {
        float dp[3], dn[3];
        if (interior) {
            for (int d = 0; d < 3; ++d) {
                dp[d] = P[t][d] - P[t - 1][d];
                dn[d] = P[t + 1][d] - P[t][d];
            }
        }
        __syncthreads();
        if (interior) {
            float g[3];
            #pragma unroll
            for (int j = 0; j < 3; ++j) {
                float ap[3], an[3], bp[3], bn[3];
                for (int d = 0; d < 3; ++d) { ap[d]=dp[d]; an[d]=dn[d]; bp[d]=dp[d]; bn[d]=dn[d]; }
                ap[j] += eps; an[j] -= eps; bp[j] -= eps; bn[j] += eps;
                float cp = sqrtf(ap[0]*ap[0]+ap[1]*ap[1]+ap[2]*ap[2])
                         + sqrtf(an[0]*an[0]+an[1]*an[1]+an[2]*an[2]);
                float cm = sqrtf(bp[0]*bp[0]+bp[1]*bp[1]+bp[2]*bp[2])
                         + sqrtf(bn[0]*bn[0]+bn[1]*bn[1]+bn[2]*bn[2]);
                g[j] = (cp - cm) / (2.f * eps);
            }
            for (int d = 0; d < 3; ++d) {
                v[d] = mom * v[d] - step * g[d];
                p[d] += v[d];
                P[t][d] = p[d];
            }
        }
        __syncthreads();
    }
    for (int d = 0; d < 3; ++d) path[(size_t)(b * T_ + t) * 3 + d] = P[t][d];
}

// ---------------------------------------------------------------------------
// Persistent LSTM direction kernel. grid = {fwd, bwd}, block = 1024 (32 waves).
// g = pre[t] + h @ whh^T via WMMA; cell state in registers mapped onto the
// WMMA D-fragment; h kept as f16 in LDS (A-operand source for next step).
// ---------------------------------------------------------------------------
__global__ void lstm_layer_kernel(const float* __restrict__ pre_fwd,
                                  const float* __restrict__ pre_bwd,
                                  const _Float16* __restrict__ whh_fwd,
                                  const _Float16* __restrict__ whh_bwd,
                                  float* __restrict__ hout)     // (B,T,512), f@0 b@256
{
    __shared__ _Float16 h_s[B_ * HID_];          // 32 KB
    const int rev = blockIdx.x;                  // 0 = forward, 1 = backward
    const float*    pre    = rev ? pre_bwd : pre_fwd;
    const _Float16* whh    = rev ? whh_bwd : whh_fwd;
    const int       diroff = rev ? HID_ : 0;

    const int tid  = threadIdx.x;
    const int wave = tid >> 5;
    const int lane = tid & 31;
    const int l16  = lane & 15;
    const int hsel = lane >> 4;

    for (int i = tid; i < B_ * HID_; i += blockDim.x) h_s[i] = (_Float16)0.0f;
    float cst[2][8];
    #pragma unroll
    for (int st = 0; st < 2; ++st)
        #pragma unroll
        for (int r = 0; r < 8; ++r) cst[st][r] = 0.0f;
    __syncthreads();

    for (int s = 0; s < T_; ++s) {
        const int tidx = rev ? (T_ - 1 - s) : s;

        // Prefetch next timestep's pre slice (64x1024 f32): 1024 threads x 256B.
        if (s + 1 < T_) {
            const int tnext = rev ? (T_ - 2 - s) : (s + 1);
            const float* pf = pre + (size_t)((tid >> 4) * T_ + tnext) * (4 * HID_)
                                  + (tid & 15) * 64;
            __builtin_prefetch(pf, 0, 1);        // -> global_prefetch_b8
        }

        v8f acc[2][4];
        #pragma unroll
        for (int st = 0; st < 2; ++st)
            #pragma unroll
            for (int g = 0; g < 4; ++g) { v8f z = {}; acc[st][g] = z; }

        // ---- GEMM phase: all waves read h_s, accumulate 2 tiles x 4 gates
        #pragma unroll
        for (int st = 0; st < 2; ++st) {
            const int tile = 2 * wave + st;      // 0..63 over (tm 0..3, tn 0..15)
            const int tm = tile >> 4, tn = tile & 15;
            const _Float16* pa = h_s + (tm * 16 + l16) * HID_ + 8 * hsel;
            for (int k0 = 0; k0 < HID_; k0 += 32) {
                v8h alo = *(const v8h*)(pa + k0);        // ds_load_b128
                v8h ahi = *(const v8h*)(pa + k0 + 16);
                v16h a = cat_h8(alo, ahi);
                #pragma unroll
                for (int g = 0; g < 4; ++g) {
                    const _Float16* pb = whh + (size_t)(g * HID_ + tn * 16 + l16) * HID_
                                             + 8 * hsel + k0;
                    v8h blo = *(const v8h*)(pb);         // global_load_b128
                    v8h bhi = *(const v8h*)(pb + 16);
                    v16h bf = cat_h8(blo, bhi);
                    acc[st][g] = __builtin_amdgcn_wmma_f32_16x16x32_f16(
                        false, a, false, bf, (short)0, acc[st][g], false, false);
                }
            }
        }
        __syncthreads();   // all h_s reads complete before h_s is rewritten

        // ---- cell update: lane owns (b = 16*tm + r + 8*hsel, n = 16*tn + l16)
        #pragma unroll
        for (int st = 0; st < 2; ++st) {
            const int tile = 2 * wave + st;
            const int tm = tile >> 4, tn = tile & 15;
            const int n = tn * 16 + l16;
            #pragma unroll
            for (int r = 0; r < 8; ++r) {
                const int b = tm * 16 + r + 8 * hsel;
                const float* prow = pre + (size_t)(b * T_ + tidx) * (4 * HID_);
                float gi = acc[st][0][r] + prow[0 * HID_ + n];
                float gf = acc[st][1][r] + prow[1 * HID_ + n];
                float gg = acc[st][2][r] + prow[2 * HID_ + n];
                float go = acc[st][3][r] + prow[3 * HID_ + n];
                float c  = sig_f(gf) * cst[st][r] + sig_f(gi) * tanhf(gg);
                cst[st][r] = c;
                float h = sig_f(go) * tanhf(c);
                h_s[b * HID_ + n] = (_Float16)h;
                hout[(size_t)(b * T_ + tidx) * (2 * HID_) + diroff + n] = h;
            }
        }
        __syncthreads();
    }
}

// ---------------------------------------------------------------------------
// Direct convolutions (NHWC input / NCHW intermediate), stride 2, pad 1, gelu.
// ---------------------------------------------------------------------------
__global__ void conv1_kernel(const float* __restrict__ in,   // (B,128,128,10) NHWC
                             const float* __restrict__ w,    // (32,10,3,3)
                             const float* __restrict__ bias,
                             float* __restrict__ out)        // (B,32,64,64) NCHW
{
    int idx = blockIdx.x * blockDim.x + threadIdx.x;
    if (idx >= B_ * 32 * 64 * 64) return;
    int x = idx & 63, y = (idx >> 6) & 63, o = (idx >> 12) & 31, b = idx >> 17;
    float s = bias[o];
    for (int ky = 0; ky < 3; ++ky) {
        int iy = 2 * y + ky - 1;
        if (iy < 0 || iy >= H_) continue;
        for (int kx = 0; kx < 3; ++kx) {
            int ix = 2 * x + kx - 1;
            if (ix < 0 || ix >= W_) continue;
            const float* ip = in + ((size_t)(b * H_ + iy) * W_ + ix) * 10;
            const float* wp = w + (o * 10 * 3 + ky) * 3 + kx;
            for (int i = 0; i < 10; ++i) s += ip[i] * wp[i * 9];
        }
    }
    out[idx] = gelu_f(s);
}

__global__ void conv2_kernel(const float* __restrict__ in,   // (B,32,64,64) NCHW
                             const float* __restrict__ w,    // (64,32,3,3)
                             const float* __restrict__ bias,
                             float* __restrict__ out)        // (B,64,32,32) NCHW
{
    int idx = blockIdx.x * blockDim.x + threadIdx.x;
    if (idx >= B_ * 64 * 32 * 32) return;
    int x = idx & 31, y = (idx >> 5) & 31, o = (idx >> 10) & 63, b = idx >> 16;
    float s = bias[o];
    for (int i = 0; i < 32; ++i) {
        for (int ky = 0; ky < 3; ++ky) {
            int iy = 2 * y + ky - 1;
            if (iy < 0 || iy >= 64) continue;
            for (int kx = 0; kx < 3; ++kx) {
                int ix = 2 * x + kx - 1;
                if (ix < 0 || ix >= 64) continue;
                s += in[((size_t)(b * 32 + i) * 64 + iy) * 64 + ix]
                   * w[((o * 32 + i) * 3 + ky) * 3 + kx];
            }
        }
    }
    out[idx] = gelu_f(s);
}

__global__ void field_mean_kernel(const float* __restrict__ in, float* __restrict__ out)
{   // (B,64,32,32) -> (B,64)
    int idx = blockIdx.x * blockDim.x + threadIdx.x;
    if (idx >= B_ * 64) return;
    float s = 0.f;
    const float* p = in + (size_t)idx * 1024;
    for (int i = 0; i < 1024; ++i) s += p[i];
    out[idx] = s * (1.0f / 1024.0f);
}

__global__ void pool_t_kernel(const float* __restrict__ h1, float* __restrict__ out)
{   // (B,T,512) -> (B,512) mean over T
    int idx = blockIdx.x * blockDim.x + threadIdx.x;
    if (idx >= B_ * 512) return;
    int b = idx / 512, k = idx % 512;
    float s = 0.f;
    for (int t = 0; t < T_; ++t) s += h1[(size_t)(b * T_ + t) * 512 + k];
    out[idx] = s * (1.0f / T_);
}

__global__ void comb_kernel(const float* __restrict__ tpool,
                            const float* __restrict__ field,
                            float* __restrict__ comb)
{   // (B,512)+(B,64) -> (B,576)
    int idx = blockIdx.x * blockDim.x + threadIdx.x;
    if (idx >= B_ * 576) return;
    int b = idx / 576, j = idx % 576;
    comb[idx] = (j < 512) ? tpool[b * 512 + j] : field[b * 64 + (j - 512)];
}

__global__ void refine_combine_kernel(const float* __restrict__ path,
                                      const float* __restrict__ r2,  // (B,192)
                                      float* __restrict__ traj)
{
    int idx = blockIdx.x * blockDim.x + threadIdx.x;
    if (idx >= B_ * T_ * 3) return;
    int b = idx / (T_ * 3);
    int r = idx % (T_ * 3);
    int t = r / 3;
    float v = path[idx] + 0.1f * r2[b * (T_ * 3) + r];
    if (t == 0 || t == T_ - 1) v = path[idx];
    traj[idx] = v;
}

// ---------------------------------------------------------------------------
// Physics: 3 rounds of velocity/accel clamping + cumsum. Block per batch.
// ---------------------------------------------------------------------------
__global__ void physics_kernel(float* __restrict__ traj)
{
    __shared__ float C[T_][3], V[T_][3], A[T_][3];
    const int b = blockIdx.x, t = threadIdx.x;
    const float DT = 0.1f;
    for (int d = 0; d < 3; ++d) C[t][d] = traj[(size_t)(b * T_ + t) * 3 + d];
    __syncthreads();
    for (int it = 0; it < 3; ++it) {
        if (t < T_ - 1) {
            float v[3], n2 = 0.f;
            for (int d = 0; d < 3; ++d) { v[d] = (C[t+1][d] - C[t][d]) / DT; n2 += v[d]*v[d]; }
            float sc = fminf(1.0f, 2.0f / (sqrtf(n2) + 1e-8f));
            for (int d = 0; d < 3; ++d) V[t][d] = v[d] * sc;
        }
        __syncthreads();
        if (t == 0) {
            float run[3] = {C[0][0], C[0][1], C[0][2]};
            for (int s = 0; s < T_ - 1; ++s)
                for (int d = 0; d < 3; ++d) { run[d] += V[s][d] * DT; C[s+1][d] = run[d]; }
        }
        __syncthreads();
        if (t < T_ - 2) {
            float a[3], n2 = 0.f;
            for (int d = 0; d < 3; ++d) { a[d] = (V[t+1][d] - V[t][d]) / DT; n2 += a[d]*a[d]; }
            float sc = fminf(1.0f, 5.0f / (sqrtf(n2) + 1e-8f));
            for (int d = 0; d < 3; ++d) A[t][d] = a[d] * sc;
        }
        __syncthreads();
        if (t == 0) {
            float run[3] = {V[0][0], V[0][1], V[0][2]};
            for (int s = 0; s < T_ - 2; ++s)
                for (int d = 0; d < 3; ++d) { run[d] += A[s][d] * DT; V[s+1][d] = run[d]; }
            float rc[3] = {C[0][0], C[0][1], C[0][2]};
            for (int s = 0; s < T_ - 1; ++s)
                for (int d = 0; d < 3; ++d) { rc[d] += V[s][d] * DT; C[s+1][d] = rc[d]; }
        }
        __syncthreads();
    }
    for (int d = 0; d < 3; ++d) traj[(size_t)(b * T_ + t) * 3 + d] = C[t][d];
}

__global__ void collision_kernel(const float* __restrict__ traj,
                                 const float* __restrict__ aff,
                                 float* __restrict__ valid)
{
    __shared__ int ok;
    const int b = blockIdx.x, t = threadIdx.x;
    if (t == 0) ok = 1;
    __syncthreads();
    float px = traj[(size_t)(b * T_ + t) * 3 + 0];
    float py = traj[(size_t)(b * T_ + t) * 3 + 1];
    int gi = (int)((px + 10.0f) / 20.0f * (float)H_);
    int gj = (int)((py + 10.0f) / 20.0f * (float)W_);
    bool inb = (gi >= 0) && (gi < H_) && (gj >= 0) && (gj < W_);
    int ii = min(max(gi, 0), H_ - 1);
    int jj = min(max(gj, 0), W_ - 1);
    float mass = 0.f;
    const float* ap = aff + ((size_t)(b * H_ + ii) * W_ + jj) * CA_;
    for (int c = 0; c < CA_; ++c) mass += ap[c];
    if (!(inb && mass <= 100.0f)) atomicAnd(&ok, 0);
    __syncthreads();
    if (t == 0) valid[b] = ok ? 1.0f : 0.0f;
}

// ---------------------------------------------------------------------------
// Host launcher
// ---------------------------------------------------------------------------
static inline void gemm(const float* X, const float* Wt, const float* b1, const float* b2,
                        float* Y, int M, int N, int K, int act, hipStream_t s)
{
    int waves  = (M / 16) * ((N + 15) / 16);
    int blocks = (waves + 3) / 4;                // 4 waves / 128-thread block
    wmma_gemm_kernel<<<dim3(blocks), dim3(128), 0, s>>>(X, Wt, b1, b2, Y, M, N, K, act);
}

extern "C" void kernel_launch(void* const* d_in, const int* in_sizes, int n_in,
                              void* d_out, int out_size, void* d_ws, size_t ws_size,
                              hipStream_t stream)
{
    (void)in_sizes; (void)n_in; (void)out_size; (void)ws_size;
    // --- inputs in setup_inputs() insertion order -------------------------
    const float* gr     = (const float*)d_in[0];   // (64,128,128,10)
    const float* start  = (const float*)d_in[1];   // (64,3)
    const float* gf     = (const float*)d_in[2];   // (64,32,768)
    const float* aff    = (const float*)d_in[3];   // (64,128,128,16)
    const float* ge_w0  = (const float*)d_in[4];   const float* ge_b0 = (const float*)d_in[5];
    const float* ge_w1  = (const float*)d_in[6];   const float* ge_b1 = (const float*)d_in[7];
    const float* ge_w2  = (const float*)d_in[8];   const float* ge_b2 = (const float*)d_in[9];
    const float* ge_w3  = (const float*)d_in[10];  const float* ge_b3 = (const float*)d_in[11];
    const float* step_p = (const float*)d_in[12];
    const float* mom_p  = (const float*)d_in[13];
    const float* l0f_wih=(const float*)d_in[14]; const float* l0f_whh=(const float*)d_in[15];
    const float* l0f_bih=(const float*)d_in[16]; const float* l0f_bhh=(const float*)d_in[17];
    const float* l0b_wih=(const float*)d_in[18]; const float* l0b_whh=(const float*)d_in[19];
    const float* l0b_bih=(const float*)d_in[20]; const float* l0b_bhh=(const float*)d_in[21];
    const float* l1f_wih=(const float*)d_in[22]; const float* l1f_whh=(const float*)d_in[23];
    const float* l1f_bih=(const float*)d_in[24]; const float* l1f_bhh=(const float*)d_in[25];
    const float* l1b_wih=(const float*)d_in[26]; const float* l1b_whh=(const float*)d_in[27];
    const float* l1b_bih=(const float*)d_in[28]; const float* l1b_bhh=(const float*)d_in[29];
    const float* cv_w0  = (const float*)d_in[30];  const float* cv_b0 = (const float*)d_in[31];
    const float* cv_w1  = (const float*)d_in[32];  const float* cv_b1 = (const float*)d_in[33];
    const float* rf_w0  = (const float*)d_in[34];  const float* rf_b0 = (const float*)d_in[35];
    const float* rf_w1  = (const float*)d_in[36];  const float* rf_b1 = (const float*)d_in[37];
    const float* rf_w2  = (const float*)d_in[38];  const float* rf_b2 = (const float*)d_in[39];
    const float* sc_w0  = (const float*)d_in[40];  const float* sc_b0 = (const float*)d_in[41];
    const float* sc_w1  = (const float*)d_in[42];  const float* sc_b1 = (const float*)d_in[43];
    const float* sc_w2  = (const float*)d_in[44];  const float* sc_b2 = (const float*)d_in[45];

    // --- workspace layout (floats) ---------------------------------------
    float* ws     = (float*)d_ws;
    float* gmean  = ws;                        // 49152
    float* x1     = gmean  + 49152;            // 32768
    float* x2     = x1     + 32768;            // 16384
    float* x3     = x2     + 16384;            // 8192
    float* genc   = x3     + 8192;             // 16384
    float* path   = genc   + 16384;            // 12288
    float* pre_f  = path   + 12288;            // 4194304
    float* pre_b  = pre_f  + 4194304;          // 4194304
    float* h0     = pre_b  + 4194304;          // 2097152
    float* h1     = h0     + 2097152;          // 2097152
    _Float16* whh16 = (_Float16*)(h1 + 2097152); // 4*262144 halves (= 524288 floats)
    float* conv2o = h1 + 2097152 + 524288;     // 4194304
    float* field  = conv2o + 4194304;          // 4096
    float* tpool  = field  + 4096;             // 32768
    float* comb   = tpool  + 32768;            // 36864
    float* r0     = comb   + 36864;            // 16384
    float* r1     = r0     + 16384;            // 16384
    float* r2     = r1     + 16384;            // 12288
    float* path32 = r2     + 12288;            // 131072 (path zero-padded K=3->32)
    float* wih0pf = path32 + 131072;           // 32768  (l0f_wih padded to (1024,32))
    float* wih0pb = wih0pf + 32768;            // 32768
    float* conv1o = pre_f;                     // alias: pre_f+pre_b free after lstm1

    float* out_traj  = (float*)d_out;          // 12288
    float* out_valid = out_traj + B_ * T_ * 3; // 64
    float* out_score = out_valid + B_;         // 64

    // 1. goal mean over L
    mean_l_kernel<<<dim3((B_*EMB_ + 255)/256), dim3(256), 0, stream>>>(gf, gmean);
    // 2. goal encoder MLP (WMMA)
    gemm(gmean, ge_w0, ge_b0, nullptr, x1,   B_, 512, EMB_, 1, stream);
    gemm(x1,    ge_w1, ge_b1, nullptr, x2,   B_, 256, 512,  1, stream);
    gemm(x2,    ge_w2, ge_b2, nullptr, x3,   B_, 128, 256,  1, stream);
    gemm(x3,    ge_w3, ge_b3, nullptr, genc, B_, 256, 128,  2, stream);
    // 3. geodesic optimization
    geodesic_kernel<<<dim3(B_), dim3(T_), 0, stream>>>(start, genc, step_p, mom_p, path);
    // 4. LSTM layer 0 input projections: zero-pad K=3 -> 32, then fast GEMM
    pad_k_kernel<<<dim3((B_*T_*32 + 255)/256), dim3(256), 0, stream>>>(path, path32, B_*T_, 3);
    pad_k_kernel<<<dim3((1024*32 + 255)/256), dim3(256), 0, stream>>>(l0f_wih, wih0pf, 1024, 3);
    pad_k_kernel<<<dim3((1024*32 + 255)/256), dim3(256), 0, stream>>>(l0b_wih, wih0pb, 1024, 3);
    gemm(path32, wih0pf, l0f_bih, l0f_bhh, pre_f, B_*T_, 1024, 32, 0, stream);
    gemm(path32, wih0pb, l0b_bih, l0b_bhh, pre_b, B_*T_, 1024, 32, 0, stream);
    // 5. convert recurrent weights to f16 once
    cvt_f16_kernel<<<dim3(1024), dim3(256), 0, stream>>>(l0f_whh, whh16 + 0*262144, 262144);
    cvt_f16_kernel<<<dim3(1024), dim3(256), 0, stream>>>(l0b_whh, whh16 + 1*262144, 262144);
    cvt_f16_kernel<<<dim3(1024), dim3(256), 0, stream>>>(l1f_whh, whh16 + 2*262144, 262144);
    cvt_f16_kernel<<<dim3(1024), dim3(256), 0, stream>>>(l1b_whh, whh16 + 3*262144, 262144);
    // 6. LSTM layer 0 recurrence (fwd+bwd, persistent workgroups)
    lstm_layer_kernel<<<dim3(2), dim3(1024), 0, stream>>>(
        pre_f, pre_b, whh16 + 0*262144, whh16 + 1*262144, h0);
    // 7. LSTM layer 1 input projections (K=512)
    gemm(h0, l1f_wih, l1f_bih, l1f_bhh, pre_f, B_*T_, 1024, 512, 0, stream);
    gemm(h0, l1b_wih, l1b_bih, l1b_bhh, pre_b, B_*T_, 1024, 512, 0, stream);
    // 8. LSTM layer 1 recurrence
    lstm_layer_kernel<<<dim3(2), dim3(1024), 0, stream>>>(
        pre_f, pre_b, whh16 + 2*262144, whh16 + 3*262144, h1);
    // 9. pool h1 over T
    pool_t_kernel<<<dim3((B_*512 + 255)/256), dim3(256), 0, stream>>>(h1, tpool);
    // 10. conv path (pre_f/pre_b are free now; conv1o aliases them)
    conv1_kernel<<<dim3((B_*32*64*64 + 255)/256), dim3(256), 0, stream>>>(gr, cv_w0, cv_b0, conv1o);
    conv2_kernel<<<dim3((B_*64*32*32 + 255)/256), dim3(256), 0, stream>>>(conv1o, cv_w1, cv_b1, conv2o);
    field_mean_kernel<<<dim3((B_*64 + 255)/256), dim3(256), 0, stream>>>(conv2o, field);
    // 11. refine MLP
    comb_kernel<<<dim3((B_*576 + 255)/256), dim3(256), 0, stream>>>(tpool, field, comb);
    gemm(comb, rf_w0, rf_b0, nullptr, r0, B_, 256, 576, 1, stream);
    gemm(r0,   rf_w1, rf_b1, nullptr, r1, B_, 256, 256, 1, stream);
    gemm(r1,   rf_w2, rf_b2, nullptr, r2, B_, 192, 256, 0, stream);
    // 12. refined traj (lin-interp is identity since r has T rows already)
    refine_combine_kernel<<<dim3((B_*T_*3 + 255)/256), dim3(256), 0, stream>>>(path, r2, out_traj);
    // 13. physics projection + collision check
    physics_kernel<<<dim3(B_), dim3(T_), 0, stream>>>(out_traj);
    collision_kernel<<<dim3(B_), dim3(T_), 0, stream>>>(out_traj, aff, out_valid);
    // 14. score MLP (reuse x1/x2)
    gemm(genc, sc_w0, sc_b0, nullptr, x1,        B_, 128, 256, 1, stream);
    gemm(x1,   sc_w1, sc_b1, nullptr, x2,        B_, 64,  128, 1, stream);
    gemm(x2,   sc_w2, sc_b2, nullptr, out_score, B_, 1,   64,  3, stream);
}
